// SetTransformer_89498528514391
// MI455X (gfx1250) — compile-verified, matrix operand order checked by reading, not run on hardware
//
#include <hip/hip_runtime.h>
#include <hip/hip_bf16.h>

// ---------------------------------------------------------------------------
// Set Transformer (linear attention) for MI455X / gfx1250.
// f16-resident activations + f16 transposed weights; GEMM tiles staged with
// the Tensor Data Mover (tensor_load_to_lds, TENSORcnt), math on
// V_WMMA_F32_16X16X32_F16 with f32 accumulation.
// ---------------------------------------------------------------------------

typedef __attribute__((ext_vector_type(16))) _Float16     v16h;
typedef __attribute__((ext_vector_type(8)))  float        v8f;
typedef __attribute__((ext_vector_type(4)))  unsigned int u32x4;
typedef __attribute__((ext_vector_type(8)))  int          i32x8;
typedef __attribute__((ext_vector_type(4)))  int          i32x4;

#define DMODEL 512
#define NHEAD  8
#define DH     64
#define BATCH  4
#define HBATCH (BATCH * NHEAD)  // 32

// K index packed into fragment VGPR v (2 halves per VGPR), ISA 7.12.2 wave32:
// VGPR0..3: K = 2v,2v+1 (lanes 0-15) / +8 (lanes 16-31); VGPR4..7: +16.
__device__ __forceinline__ int kb16(int v, int hi) {
  return 2 * (v & 3) + ((v & 4) ? 16 : 0) + (hi ? 8 : 0);
}

__device__ __forceinline__ v8f wmma_f16(v16h a, v16h b, v8f c) {
  return __builtin_amdgcn_wmma_f32_16x16x32_f16(false, a, false, b, (short)0, c,
                                                false, false);
}

// ---------------------------------------------------------------------------
// TDM: 2D tile (tile_w halves x tile_h rows) global->LDS, row stride
// stride_elems halves, LDS padded +8 halves (4 DWORDs) per 32 halves (16 DW).
// Descriptor packing per CDNA5 ISA ch.8 (D# groups 0/1; groups 2/3 zero = 2D).
// ---------------------------------------------------------------------------
__device__ __forceinline__ void tdm_load_2d(unsigned lds_off, const void* gptr,
                                            unsigned tile_w, unsigned tile_h,
                                            unsigned stride_elems,
                                            unsigned tdim0, unsigned tdim1) {
  unsigned long long ga = (unsigned long long)(size_t)gptr;
  u32x4 g0;
  g0.x = 1u;  // count=1, is_restore=0, gather_mode=0
  g0.y = lds_off;
  g0.z = (unsigned)(ga & 0xffffffffull);
  g0.w = (unsigned)((ga >> 32) & 0x01ffffffull) | (2u << 30);  // type=2 image
  i32x8 g1;
  // data_size=1 (2B) | pad_enable | pad_interval=3 (16 DW) | pad_amount=3 (4 DW)
  g1[0] = (int)((1u << 16) | (1u << 20) | (3u << 22) | (3u << 25));
  g1[1] = (int)((tdim0 & 0xffffu) << 16);  // abar_addr=0 | tensor_dim0 lo16
  g1[2] = (int)(((tdim0 >> 16) & 0xffffu) | ((tdim1 & 0xffffu) << 16));
  g1[3] = (int)(((tdim1 >> 16) & 0xffffu) | ((tile_w & 0xffffu) << 16));
  g1[4] = (int)(tile_h & 0xffffu);  // tile_dim1 | tile_dim2=0
  g1[5] = (int)stride_elems;        // tensor_dim0_stride lo32
  g1[6] = 0;                        // stride hi | dim1_stride lo (unused, 2D)
  g1[7] = 0;
  i32x4 z4 = {0, 0, 0, 0};
#if __clang_major__ >= 23
  i32x8 z8 = {0, 0, 0, 0, 0, 0, 0, 0};
  __builtin_amdgcn_tensor_load_to_lds(g0, g1, z4, z4, z8, 0);
#else
  __builtin_amdgcn_tensor_load_to_lds(g0, g1, z4, z4, 0);
#endif
}

// ---------------------------------------------------------------------------
// GEMM: C[M,N] = act(A[M,K] @ Wt[N,K]^T + bias) (+ res), A/Wt f16, acc f32.
// act: 0 none, 1 relu, 2 elu(x)+1. Outputs: C16 (f16) and/or C32 (f32).
// M%128==0, N%64==0, K%32==0. Block 256 thr = 8 waves (4M x 2N), 32x32/wave.
// Tiles staged by TDM, double buffered; zero staging VALU/DS-store traffic.
// ---------------------------------------------------------------------------
#define BM 128
#define BN 64
#define BK 32
#define LDT 40

__global__ __launch_bounds__(256) void gemm16(
    const _Float16* __restrict__ A, const _Float16* __restrict__ Wt,
    const float* __restrict__ bias, const _Float16* __restrict__ res,
    _Float16* __restrict__ C16, float* __restrict__ C32, int M, int K, int N,
    int act) {
  __shared__ _Float16 As[2][BM][LDT];
  __shared__ _Float16 Bs[2][BN][LDT];
  const int tid = threadIdx.x;
  const int lane = tid & 31, wave = tid >> 5;
  const int hi = lane >> 4, ml = lane & 15;
  const int wm = wave >> 1;  // 0..3 -> 32 rows
  const int wn = wave & 1;   // 0..1 -> 32 cols
  const int bm = blockIdx.y * BM, bn = blockIdx.x * BN;

  v8f c00 = {}, c01 = {}, c10 = {}, c11 = {};

  if (tid < 32) {  // wave 0 drives the TDM
    tdm_load_2d((unsigned)(size_t)&As[0][0][0], A + (size_t)bm * K, BK, BM, K,
                K, M);
    tdm_load_2d((unsigned)(size_t)&Bs[0][0][0], Wt + (size_t)bn * K, BK, BN, K,
                K, N);
  }
  int p = 0;
  for (int k0 = 0; k0 < K; k0 += BK) {
    if (tid < 32) {
      if (k0 + BK < K) {
        tdm_load_2d((unsigned)(size_t)&As[p ^ 1][0][0],
                    A + (size_t)bm * K + k0 + BK, BK, BM, K, K, M);
        tdm_load_2d((unsigned)(size_t)&Bs[p ^ 1][0][0],
                    Wt + (size_t)bn * K + k0 + BK, BK, BN, K, K, N);
        __builtin_amdgcn_s_wait_tensorcnt(2);  // oldest pair (buf p) done
      } else {
        __builtin_amdgcn_s_wait_tensorcnt(0);
      }
    }
    __syncthreads();  // buf p visible to all waves

    v16h a0, a1, b0, b1;
#pragma unroll
    for (int v = 0; v < 8; ++v) {
      const int kk = kb16(v, hi);
      a0[2 * v]     = As[p][wm * 32 + ml][kk];
      a0[2 * v + 1] = As[p][wm * 32 + ml][kk + 1];
      a1[2 * v]     = As[p][wm * 32 + 16 + ml][kk];
      a1[2 * v + 1] = As[p][wm * 32 + 16 + ml][kk + 1];
      b0[2 * v]     = Bs[p][wn * 32 + ml][kk];
      b0[2 * v + 1] = Bs[p][wn * 32 + ml][kk + 1];
      b1[2 * v]     = Bs[p][wn * 32 + 16 + ml][kk];
      b1[2 * v + 1] = Bs[p][wn * 32 + 16 + ml][kk + 1];
    }
    c00 = wmma_f16(a0, b0, c00);
    c01 = wmma_f16(a0, b1, c01);
    c10 = wmma_f16(a1, b0, c10);
    c11 = wmma_f16(a1, b1, c11);
    __syncthreads();  // reads of buf p done before next-iter TDM reuses it
    p ^= 1;
  }

  const int n0 = bn + wn * 32 + ml;
  const int n1 = n0 + 16;
  const float bv0 = bias ? bias[n0] : 0.f;
  const float bv1 = bias ? bias[n1] : 0.f;
  const v8f* cs[4] = {&c00, &c01, &c10, &c11};
#pragma unroll
  for (int t = 0; t < 4; ++t) {
    const int i = t >> 1, j = t & 1;
    const int n = j ? n1 : n0;
    const float bv = j ? bv1 : bv0;
#pragma unroll
    for (int v = 0; v < 8; ++v) {
      const int m = bm + wm * 32 + i * 16 + v + 8 * hi;
      float x = (*cs[t])[v] + bv;
      if (act == 1) x = fmaxf(x, 0.f);
      else if (act == 2) x = (x > 0.f) ? x + 1.f : __expf(x);
      if (res) x += (float)res[(size_t)m * N + n];
      if (C16) C16[(size_t)m * N + n] = (_Float16)x;
      if (C32) C32[(size_t)m * N + n] = x;
    }
  }
}

// ---------------------------------------------------------------------------
// kv_reduce: per (b,h): KV[d][m] = sum_s K[s,d]*V[s,m]; Ksum[d] = sum_s K[s,d]
// Inputs f16 [B*S,512]. LDS stores contiguous b128 (scatter on global side).
// ---------------------------------------------------------------------------
__global__ __launch_bounds__(256) void kv_reduce(
    const _Float16* __restrict__ Km, const _Float16* __restrict__ Vm,
    float* __restrict__ KVo, float* __restrict__ Kso, int S) {
  __shared__ _Float16 Kt[64][40];  // [d][s]
  __shared__ _Float16 Vt[64][40];  // [m][s]
  const int bh = blockIdx.x;
  const int b = bh / NHEAD, h = bh % NHEAD;
  const int tid = threadIdx.x;
  const int lane = tid & 31, wave = tid >> 5;
  const int hi = lane >> 4, ml = lane & 15;
  const int wm = wave >> 1, wn = wave & 1;
  const int ld = tid & 63;         // d / m
  const int sc = (tid >> 6) * 8;   // s chunk of 8

  float ksum = 0.f;
  v8f c0 = {}, c1 = {};

  for (int s0 = 0; s0 < S; s0 += 32) {
#pragma unroll
    for (int j = 0; j < 8; ++j) {
      const size_t row = (size_t)(b * S + s0 + sc + j) * DMODEL + h * DH + ld;
      Kt[ld][sc + j] = Km[row];
      Vt[ld][sc + j] = Vm[row];
    }
    __syncthreads();
    if (tid < 64) {
#pragma unroll
      for (int s = 0; s < 32; ++s) ksum += (float)Kt[tid][s];
    }
    v16h af, b0, b1;
#pragma unroll
    for (int v = 0; v < 8; ++v) {
      const int kk = kb16(v, hi);
      af[2 * v]     = Kt[wm * 16 + ml][kk];
      af[2 * v + 1] = Kt[wm * 16 + ml][kk + 1];
      b0[2 * v]     = Vt[wn * 32 + ml][kk];
      b0[2 * v + 1] = Vt[wn * 32 + ml][kk + 1];
      b1[2 * v]     = Vt[wn * 32 + 16 + ml][kk];
      b1[2 * v + 1] = Vt[wn * 32 + 16 + ml][kk + 1];
    }
    c0 = wmma_f16(af, b0, c0);
    c1 = wmma_f16(af, b1, c1);
    __syncthreads();
  }
#pragma unroll
  for (int v = 0; v < 8; ++v) {
    const int d = wm * 16 + v + 8 * hi;
    const int n = wn * 32 + ml;
    KVo[(size_t)bh * 4096 + d * 64 + n]      = c0[v];
    KVo[(size_t)bh * 4096 + d * 64 + n + 16] = c1[v];
  }
  if (tid < 64) Kso[bh * 64 + tid] = ksum;
}

// ---------------------------------------------------------------------------
// attn_out: O[l][m] = (sum_d Q[l,d]*KV[d][m]) / (Q[l,:].Ksum + 1e-6), f16 out.
// ---------------------------------------------------------------------------
__global__ __launch_bounds__(256) void attn_out(
    const _Float16* __restrict__ Qm, const float* __restrict__ KVws,
    const float* __restrict__ Ksws, _Float16* __restrict__ Om, int Lq) {
  __shared__ _Float16 Qs[64][72];  // [l][d]
  __shared__ _Float16 Bs[64][72];  // [n=m][k=d]
  __shared__ float ks_s[64];
  __shared__ float zs[64];
  const int bh = blockIdx.x;
  const int b = bh / NHEAD, h = bh % NHEAD;
  const int row0 = b * Lq + blockIdx.y * 64;
  const int tid = threadIdx.x;
  const int lane = tid & 31, wave = tid >> 5;
  const int hi = lane >> 4, ml = lane & 15;
  const int wm = wave >> 1, wn = wave & 1;

  {
    const int l = tid >> 2, d0 = (tid & 3) * 16;
    const _Float16* qp = Qm + (size_t)(row0 + l) * DMODEL + h * DH + d0;
#pragma unroll
    for (int j = 0; j < 16; ++j) Qs[l][d0 + j] = qp[j];
    const int n = tid & 63, kc = (tid >> 6) * 16;
#pragma unroll
    for (int j = 0; j < 16; ++j)
      Bs[n][kc + j] = (_Float16)KVws[(size_t)bh * 4096 + (kc + j) * 64 + n];
  }
  if (tid < 64) ks_s[tid] = Ksws[bh * 64 + tid];
  __syncthreads();
  if (tid < 64) {
    float acc = 0.f;
#pragma unroll
    for (int d = 0; d < 64; ++d) acc += (float)Qs[tid][d] * ks_s[d];
    zs[tid] = 1.f / (acc + 1e-6f);
  }
  __syncthreads();

  v8f c0 = {}, c1 = {};
#pragma unroll
  for (int k0 = 0; k0 < 64; k0 += 32) {
    v16h af, b0, b1;
#pragma unroll
    for (int v = 0; v < 8; ++v) {
      const int kk = k0 + kb16(v, hi);
      af[2 * v]     = Qs[wm * 16 + ml][kk];
      af[2 * v + 1] = Qs[wm * 16 + ml][kk + 1];
      b0[2 * v]     = Bs[wn * 32 + ml][kk];
      b0[2 * v + 1] = Bs[wn * 32 + ml][kk + 1];
      b1[2 * v]     = Bs[wn * 32 + 16 + ml][kk];
      b1[2 * v + 1] = Bs[wn * 32 + 16 + ml][kk + 1];
    }
    c0 = wmma_f16(af, b0, c0);
    c1 = wmma_f16(af, b1, c1);
  }
#pragma unroll
  for (int v = 0; v < 8; ++v) {
    const int mr = wm * 16 + v + 8 * hi;
    const float z = zs[mr];
    const int n = wn * 32 + ml;
    Om[(size_t)(row0 + mr) * DMODEL + h * DH + n]      = (_Float16)(c0[v] * z);
    Om[(size_t)(row0 + mr) * DMODEL + h * DH + n + 16] = (_Float16)(c1[v] * z);
  }
}

// ---------------------------------------------------------------------------
// LayerNorm over D=512, f16 in/out, f32 math (biased var, eps=1e-5).
// ---------------------------------------------------------------------------
__global__ __launch_bounds__(256) void layernorm16(
    const _Float16* __restrict__ in, _Float16* __restrict__ out,
    const float* __restrict__ g, const float* __restrict__ be) {
  __shared__ float r1[256], r2[256];
  const int row = blockIdx.x, tid = threadIdx.x;
  const float x0 = (float)in[(size_t)row * DMODEL + tid];
  const float x1 = (float)in[(size_t)row * DMODEL + 256 + tid];
  r1[tid] = x0 + x1;
  r2[tid] = x0 * x0 + x1 * x1;
  __syncthreads();
  for (int s = 128; s > 0; s >>= 1) {
    if (tid < s) { r1[tid] += r1[tid + s]; r2[tid] += r2[tid + s]; }
    __syncthreads();
  }
  const float mean = r1[0] * (1.f / 512.f);
  const float var = r2[0] * (1.f / 512.f) - mean * mean;
  const float rs = rsqrtf(var + 1e-5f);
  out[(size_t)row * DMODEL + tid] =
      (_Float16)((x0 - mean) * rs * g[tid] + be[tid]);
  out[(size_t)row * DMODEL + 256 + tid] =
      (_Float16)((x1 - mean) * rs * g[256 + tid] + be[256 + tid]);
}

// seeds f32 [256,512] -> Sp16 f16 [1024,512]
__global__ void seed_bcast(const float* __restrict__ seeds,
                           _Float16* __restrict__ Sp) {
  const int idx = blockIdx.x * blockDim.x + threadIdx.x;
  Sp[idx] = (_Float16)seeds[(((idx >> 9) & 255) << 9) + (idx & 511)];
}

// f32 -> f16 elementwise
__global__ void cvt16(const float* __restrict__ in, _Float16* __restrict__ out,
                      int n) {
  const int idx = blockIdx.x * blockDim.x + threadIdx.x;
  if (idx < n) out[idx] = (_Float16)in[idx];
}

// W[K,N] f32 -> Wt[N,K] f16 (transpose + convert), one-time per launch
__global__ void wt_cvt(const float* __restrict__ W, _Float16* __restrict__ Wt,
                       int K, int N) {
  const int idx = blockIdx.x * blockDim.x + threadIdx.x;
  if (idx >= K * N) return;
  const int n = idx % N, k = idx / N;
  Wt[(size_t)n * K + k] = (_Float16)W[idx];
}

// ---------------------------------------------------------------------------
// Host orchestration
// ---------------------------------------------------------------------------
extern "C" void kernel_launch(void* const* d_in, const int* in_sizes, int n_in,
                              void* d_out, int out_size, void* d_ws,
                              size_t ws_size, hipStream_t stream) {
  (void)in_sizes; (void)n_in; (void)out_size; (void)ws_size;
  const float* src   = (const float*)d_in[0];
  const float* W_in  = (const float*)d_in[1];
  const float* b_in  = (const float*)d_in[2];
  const float* seeds = (const float*)d_in[3];
  const float* pre_W[6]  = {(const float*)d_in[4],  (const float*)d_in[5],
                            (const float*)d_in[6],  (const float*)d_in[7],
                            (const float*)d_in[8],  (const float*)d_in[9]};
  const float* pre_b[6]  = {(const float*)d_in[10], (const float*)d_in[11],
                            (const float*)d_in[12], (const float*)d_in[13],
                            (const float*)d_in[14], (const float*)d_in[15]};
  const float* pre_be1 = (const float*)d_in[16];
  const float* pre_be2 = (const float*)d_in[17];
  const float* pre_bf  = (const float*)d_in[18];
  const float* pre_g1  = (const float*)d_in[19];
  const float* pre_g2  = (const float*)d_in[20];
  const float* pre_gf  = (const float*)d_in[21];
  const float* post_W[6] = {(const float*)d_in[22], (const float*)d_in[23],
                            (const float*)d_in[24], (const float*)d_in[25],
                            (const float*)d_in[26], (const float*)d_in[27]};
  const float* post_b[6] = {(const float*)d_in[28], (const float*)d_in[29],
                            (const float*)d_in[30], (const float*)d_in[31],
                            (const float*)d_in[32], (const float*)d_in[33]};
  const float* post_be1 = (const float*)d_in[34];
  const float* post_be2 = (const float*)d_in[35];
  const float* post_bf  = (const float*)d_in[36];
  const float* post_g1  = (const float*)d_in[37];
  const float* post_g2  = (const float*)d_in[38];
  const float* post_gf  = (const float*)d_in[39];
  const float* pool_W[4] = {(const float*)d_in[40], (const float*)d_in[41],
                            (const float*)d_in[42], (const float*)d_in[43]};
  const float* pool_b[4] = {(const float*)d_in[44], (const float*)d_in[45],
                            (const float*)d_in[46], (const float*)d_in[47]};
  const float* W_out = (const float*)d_in[48];
  const float* b_out = (const float*)d_in[49];

  // -------- workspace layout (f16 activations + f16 transposed weights) ----
  const size_t RMH = 8192ull * 512;  // one [8192,512] f16 tensor, in halves
  _Float16* X16 = (_Float16*)d_ws;           // 8 MB
  _Float16* Y16 = X16 + RMH;                 // 8 MB
  _Float16* U16 = Y16 + RMH;                 // 32 MB: Q/K/V/O or FFN hidden
  _Float16* Q16 = U16;
  _Float16* K16 = U16 + RMH;
  _Float16* V16 = U16 + 2 * RMH;
  _Float16* O16 = U16 + 3 * RMH;
  _Float16* H16 = U16;                       // [8192,2048] aliases Q..O
  float* KV = (float*)(U16 + 4 * RMH);       // [32][64][64]
  float* Ks = KV + (size_t)HBATCH * 64 * 64; // [32][64]
  _Float16* Sp16  = (_Float16*)(Ks + (size_t)HBATCH * 64);  // [1024,512]
  _Float16* Src16 = Sp16 + 1024ull * 512;                   // [8192,128]
  _Float16* WtBase = Src16 + 8192ull * 128;  // ~27 MB of f16 weights
  size_t wt_off = 0;

  auto cvtW = [&](const float* W, int K, int N) -> const _Float16* {
    _Float16* dst = WtBase + wt_off;
    wt_off += (size_t)K * N;
    const int total = K * N;
    wt_cvt<<<(total + 255) / 256, 256, 0, stream>>>(W, dst, K, N);
    return dst;
  };

  // convert all weights up front (cheap, ~14M elems)
  const _Float16* tW_in = cvtW(W_in, 128, 512);
  const _Float16 *tpre[2][6], *tpost[2][6], *tpool[4];
  for (int i = 0; i < 2; ++i)
    for (int w = 0; w < 6; ++w) {
      const int K = (w == 5) ? 2048 : 512;
      const int N = (w == 4) ? 2048 : 512;
      tpre[i][w]  = cvtW(pre_W[w]  + (size_t)i * K * N, K, N);
      tpost[i][w] = cvtW(post_W[w] + (size_t)i * K * N, K, N);
    }
  for (int w = 0; w < 4; ++w) tpool[w] = cvtW(pool_W[w], 512, 512);
  const _Float16* tW_out = cvtW(W_out, 512, 128);

  cvt16<<<(8192 * 128) / 256, 256, 0, stream>>>(src, Src16, 8192 * 128);

  auto gemm = [&](const _Float16* A, const _Float16* Wt, const float* bias,
                  const _Float16* res, _Float16* C16p, float* C32p, int M,
                  int K, int N, int act) {
    dim3 g(N / BN, M / BM);
    gemm16<<<g, 256, 0, stream>>>(A, Wt, bias, res, C16p, C32p, M, K, N, act);
  };

  auto enc_layer = [&](const _Float16* const* tw, const float* const* bb,
                       const float* g1, const float* be1, const float* g2,
                       const float* be2, int rows, int Nb) {
    gemm(X16, tw[0], bb[0], nullptr, Q16, nullptr, rows, 512, 512, 2);
    gemm(X16, tw[1], bb[1], nullptr, K16, nullptr, rows, 512, 512, 2);
    gemm(X16, tw[2], bb[2], nullptr, V16, nullptr, rows, 512, 512, 0);
    kv_reduce<<<HBATCH, 256, 0, stream>>>(K16, V16, KV, Ks, Nb);
    attn_out<<<dim3(HBATCH, Nb / 64), 256, 0, stream>>>(Q16, KV, Ks, O16, Nb);
    gemm(O16, tw[3], bb[3], X16, Y16, nullptr, rows, 512, 512, 0);
    layernorm16<<<rows, 256, 0, stream>>>(Y16, X16, g1, be1);
    gemm(X16, tw[4], bb[4], nullptr, H16, nullptr, rows, 512, 2048, 1);
    gemm(H16, tw[5], bb[5], X16, Y16, nullptr, rows, 2048, 512, 0);
    layernorm16<<<rows, 256, 0, stream>>>(Y16, X16, g2, be2);
  };

  // ---- input projection
  gemm(Src16, tW_in, b_in, nullptr, X16, nullptr, 8192, 128, 512, 0);

  // ---- pre encoder (8192 tokens, S=2048 per batch)
  for (int i = 0; i < 2; ++i) {
    const float* bb[6] = {pre_b[0] + i * 512,  pre_b[1] + i * 512,
                          pre_b[2] + i * 512,  pre_b[3] + i * 512,
                          pre_b[4] + i * 2048, pre_b[5] + i * 512};
    enc_layer(tpre[i], bb, pre_g1 + i * 512, pre_be1 + i * 512,
              pre_g2 + i * 512, pre_be2 + i * 512, 8192, 2048);
  }
  layernorm16<<<8192, 256, 0, stream>>>(X16, X16, pre_gf, pre_bf);

  // ---- pooling attention (seeds as queries)
  seed_bcast<<<(1024 * 512) / 256, 256, 0, stream>>>(seeds, Sp16);
  gemm(Sp16, tpool[0], pool_b[0], nullptr, Q16, nullptr, 1024, 512, 512, 2);
  gemm(X16, tpool[1], pool_b[1], nullptr, K16, nullptr, 8192, 512, 512, 2);
  gemm(X16, tpool[2], pool_b[2], nullptr, V16, nullptr, 8192, 512, 512, 0);
  kv_reduce<<<HBATCH, 256, 0, stream>>>(K16, V16, KV, Ks, 2048);
  attn_out<<<dim3(HBATCH, 4), 256, 0, stream>>>(Q16, KV, Ks, O16, 256);
  gemm(O16, tpool[3], pool_b[3], nullptr, X16, nullptr, 1024, 512, 512, 0);

  // ---- post encoder (1024 tokens, S=256 per batch)
  for (int i = 0; i < 2; ++i) {
    const float* bb[6] = {post_b[0] + i * 512,  post_b[1] + i * 512,
                          post_b[2] + i * 512,  post_b[3] + i * 512,
                          post_b[4] + i * 2048, post_b[5] + i * 512};
    enc_layer(tpost[i], bb, post_g1 + i * 512, post_be1 + i * 512,
              post_g2 + i * 512, post_be2 + i * 512, 1024, 256);
  }
  layernorm16<<<1024, 256, 0, stream>>>(X16, X16, post_gf, post_bf);

  // ---- output projection -> f32 d_out
  gemm(X16, tW_out, b_out, nullptr, nullptr, (float*)d_out, 1024, 512, 128, 0);
}
